// ReAttention_9637906612497
// MI455X (gfx1250) — matrix operand order repaired and checked by reading, compiled, not run on hardware
//
#include <hip/hip_runtime.h>

#define B_  4
#define N_  1024
#define C_  768
#define NH_ 12
#define HD_ 64

static constexpr float SCALE = 0.125f;   // 64^-0.5
static constexpr float EPS_  = 1e-5f;

typedef __attribute__((ext_vector_type(16))) _Float16 v16h;
typedef __attribute__((ext_vector_type(8)))  _Float16 v8h;
typedef __attribute__((ext_vector_type(8)))  float    v8f;

union H16 { v16h v; v8h p[2]; _Float16 e[16]; };
union F8  { v8f  v; float e[8]; };

// ---------------------------------------------------------------------------
// WMMA helper (CDNA5 16x16x32 f16 -> f32)
// ---------------------------------------------------------------------------
__device__ inline v8f wmma_f16(v16h a, v16h b, v8f c) {
  return __builtin_amdgcn_wmma_f32_16x16x32_f16(
      /*neg_a=*/false, a, /*neg_b=*/false, b,
      /*c_mod=*/(short)0, c, /*reuse_a=*/false, /*reuse_b=*/false);
}

// A-fragment (16x32 f16, MxK) from row-major f16 matrix, row stride ld halves.
// lane: row = lane&15, hi = lane>>4; elems 0..7 at K=8*hi+e, 8..15 at K=16+8*hi+(e-8)
__device__ inline v16h load_frag_a16(const _Float16* __restrict__ base, int ld, int lane) {
  int row = lane & 15, hi = lane >> 4;
  const _Float16* p = base + (size_t)row * ld + 8 * hi;
  H16 r;
  r.p[0] = *(const v8h*)(p);
  r.p[1] = *(const v8h*)(p + 16);
  return r.v;
}

// B-fragment (32x16 f16, KxN) where B = R^T, R row-major [N][K], row stride ld.
// lane: N = lane&15, hi = lane>>4; elems 0..7 at K=8*hi+e, 8..15 at K=16+8*hi+(e-8)
__device__ inline v16h load_frag_bt16(const _Float16* __restrict__ base, int ld, int lane) {
  const _Float16* p = base + (size_t)(lane & 15) * ld + 8 * (lane >> 4);
  H16 r;
  r.p[0] = *(const v8h*)(p);
  r.p[1] = *(const v8h*)(p + 16);
  return r.v;
}

// ---------------------------------------------------------------------------
// K0: cast proj_w (768x768 f32, [j][c]) -> f16 same layout
// ---------------------------------------------------------------------------
__global__ void k_w16cast(const float* __restrict__ w, _Float16* __restrict__ w16) {
  int i = blockIdx.x * 256 + threadIdx.x;
  if (i < C_ * C_) w16[i] = (_Float16)w[i];
}

// ---------------------------------------------------------------------------
// K1: 3x3 SAME convs -> Q,K row-major [b,h,n,d] f16 ; V transposed [b,h,d,n] f16
// one block per token image (4096 blocks x 256 threads)
// ---------------------------------------------------------------------------
__global__ void k_conv_qkv(const float* __restrict__ X,
                           const float* __restrict__ wq,
                           const float* __restrict__ wk,
                           const float* __restrict__ wv,
                           _Float16* __restrict__ Q,
                           _Float16* __restrict__ K,
                           _Float16* __restrict__ Vt) {
  __shared__ float xs[3][18][18];
  int bn = blockIdx.x;                  // b*1024 + n
  int b  = bn >> 10, n = bn & 1023;
  int t  = threadIdx.x;                 // 256 threads = one pixel each
  for (int i = t; i < 3 * 18 * 18; i += 256) ((float*)xs)[i] = 0.0f;
  __syncthreads();
  const float* xp = X + (size_t)bn * C_;
  for (int i = t; i < C_; i += 256) {
    int ic = i >> 8, rem = i & 255, y = rem >> 4, x = rem & 15;
    xs[ic][y + 1][x + 1] = xp[i];
  }
  __syncthreads();
  int y = t >> 4, x = t & 15;
#pragma unroll
  for (int oc = 0; oc < 3; ++oc) {
    float aq = 0.f, ak = 0.f, av = 0.f;
#pragma unroll
    for (int ic = 0; ic < 3; ++ic)
#pragma unroll
      for (int dy = 0; dy < 3; ++dy)
#pragma unroll
        for (int dx = 0; dx < 3; ++dx) {
          float xv = xs[ic][y + dy][x + dx];
          int wi = ((oc * 3 + ic) * 3 + dy) * 3 + dx;
          aq = fmaf(xv, wq[wi], aq);
          ak = fmaf(xv, wk[wi], ak);
          av = fmaf(xv, wv[wi], av);
        }
    int c = oc * 256 + t;               // flattened channel
    int h = c >> 6, d = c & 63;
    size_t qk = (((size_t)(b * NH_ + h)) * N_ + n) * HD_ + d;
    Q[qk] = (_Float16)aq;
    K[qk] = (_Float16)ak;
    Vt[(((size_t)(b * NH_ + h)) * HD_ + d) * N_ + n] = (_Float16)av;
  }
}

// ---------------------------------------------------------------------------
// K2: softmax row stats: rowmax and 1/rowsumexp of scale*Q.K^T per (b,h,n)
// one wave per (b,h, 16-row tile); two sweeps over m recomputing scores w/ WMMA
// ---------------------------------------------------------------------------
__global__ void k_rowstats(const _Float16* __restrict__ Q,
                           const _Float16* __restrict__ K,
                           float* __restrict__ rowmax,
                           float* __restrict__ rowinv) {
  int blk = blockIdx.x;                 // bh*64 + n16
  int n16 = blk & 63, bh = blk >> 6;
  int lane = threadIdx.x, hi = lane >> 4;
  const _Float16* qb = Q + ((size_t)bh * N_ + n16 * 16) * HD_;
  const _Float16* kb = K + (size_t)bh * N_ * HD_;
  v16h a0 = load_frag_a16(qb, HD_, lane);
  v16h a1 = load_frag_a16(qb + 32, HD_, lane);

  float m8[8];
#pragma unroll
  for (int r = 0; r < 8; ++r) m8[r] = -3.4e38f;
  for (int mt = 0; mt < 64; ++mt) {
    const _Float16* kt = kb + (size_t)mt * 16 * HD_;
    v16h b0 = load_frag_bt16(kt, HD_, lane);
    v16h b1 = load_frag_bt16(kt + 32, HD_, lane);
    F8 c; c.v = (v8f){};
    c.v = wmma_f16(a0, b0, c.v);
    c.v = wmma_f16(a1, b1, c.v);
#pragma unroll
    for (int r = 0; r < 8; ++r) m8[r] = fmaxf(m8[r], c.e[r] * SCALE);
  }
#pragma unroll
  for (int r = 0; r < 8; ++r)
#pragma unroll
    for (int off = 1; off < 16; off <<= 1)
      m8[r] = fmaxf(m8[r], __shfl_xor(m8[r], off, 16));

  float s8[8];
#pragma unroll
  for (int r = 0; r < 8; ++r) s8[r] = 0.f;
  for (int mt = 0; mt < 64; ++mt) {
    const _Float16* kt = kb + (size_t)mt * 16 * HD_;
    v16h b0 = load_frag_bt16(kt, HD_, lane);
    v16h b1 = load_frag_bt16(kt + 32, HD_, lane);
    F8 c; c.v = (v8f){};
    c.v = wmma_f16(a0, b0, c.v);
    c.v = wmma_f16(a1, b1, c.v);
#pragma unroll
    for (int r = 0; r < 8; ++r) s8[r] += __expf(c.e[r] * SCALE - m8[r]);
  }
#pragma unroll
  for (int r = 0; r < 8; ++r)
#pragma unroll
    for (int off = 1; off < 16; off <<= 1)
      s8[r] += __shfl_xor(s8[r], off, 16);

  if ((lane & 15) == 0) {
#pragma unroll
    for (int r = 0; r < 8; ++r) {
      size_t idx = (size_t)bh * N_ + n16 * 16 + r + 8 * hi;
      rowmax[idx] = m8[r];
      rowinv[idx] = 1.0f / s8[r];
    }
  }
}

// ---------------------------------------------------------------------------
// K3: recompute softmax per 16x16 tile for all 12 heads, head-mix + bias,
// write pre-BN attn to d_out, emit per-block partial sum/sumsq per head g.
// one wave per (b, n16, m16): 16384 blocks.
// ---------------------------------------------------------------------------
__global__ void k_mix(const _Float16* __restrict__ Q,
                      const _Float16* __restrict__ K,
                      const float* __restrict__ rowmax,
                      const float* __restrict__ rowinv,
                      const float* __restrict__ rw,
                      const float* __restrict__ rb,
                      float* __restrict__ attn,
                      float* __restrict__ partial) {
  int blk = blockIdx.x;
  int m16 = blk & 63, n16 = (blk >> 6) & 63, b = blk >> 12;
  int lane = threadIdx.x, hi = lane >> 4, col = lane & 15;

  float p[NH_][8];
#pragma unroll
  for (int h = 0; h < NH_; ++h) {
    int bh = b * NH_ + h;
    const _Float16* qb = Q + ((size_t)bh * N_ + n16 * 16) * HD_;
    const _Float16* kb = K + ((size_t)bh * N_ + m16 * 16) * HD_;
    v16h a0 = load_frag_a16(qb, HD_, lane);
    v16h a1 = load_frag_a16(qb + 32, HD_, lane);
    v16h b0 = load_frag_bt16(kb, HD_, lane);
    v16h b1 = load_frag_bt16(kb + 32, HD_, lane);
    F8 c; c.v = (v8f){};
    c.v = wmma_f16(a0, b0, c.v);
    c.v = wmma_f16(a1, b1, c.v);
#pragma unroll
    for (int r = 0; r < 8; ++r) {
      size_t sidx = (size_t)bh * N_ + n16 * 16 + r + 8 * hi;
      p[h][r] = __expf(c.e[r] * SCALE - rowmax[sidx]) * rowinv[sidx];
    }
  }

#pragma unroll
  for (int g = 0; g < NH_; ++g) {
    float w[NH_];
#pragma unroll
    for (int h = 0; h < NH_; ++h) w[h] = rw[g * NH_ + h];
    float bias = rb[g];
    float ssum = 0.f, ssq = 0.f;
#pragma unroll
    for (int r = 0; r < 8; ++r) {
      float a = bias;
#pragma unroll
      for (int h = 0; h < NH_; ++h) a = fmaf(w[h], p[h][r], a);
      int n = n16 * 16 + r + 8 * hi, m = m16 * 16 + col;
      attn[(((size_t)(b * NH_ + g)) * N_ + n) * N_ + m] = a;
      ssum += a; ssq = fmaf(a, a, ssq);
    }
#pragma unroll
    for (int off = 1; off < 32; off <<= 1) {
      ssum += __shfl_xor(ssum, off, 32);
      ssq  += __shfl_xor(ssq,  off, 32);
    }
    if (lane == 0) {
      partial[((size_t)(g * 2 + 0)) * 16384 + blk] = ssum;
      partial[((size_t)(g * 2 + 1)) * 16384 + blk] = ssq;
    }
  }
}

// ---------------------------------------------------------------------------
// K4: reduce partials -> per-head BN scale/shift (12 blocks x 256 threads)
// ---------------------------------------------------------------------------
__global__ void k_redstats(const float* __restrict__ partial,
                           const float* __restrict__ gamma,
                           const float* __restrict__ beta,
                           float* __restrict__ stats) {
  __shared__ float ssum[256], ssq[256];
  int g = blockIdx.x, t = threadIdx.x;
  float a = 0.f, q = 0.f;
  for (int i = t; i < 16384; i += 256) {
    a += partial[((size_t)(g * 2 + 0)) * 16384 + i];
    q += partial[((size_t)(g * 2 + 1)) * 16384 + i];
  }
  ssum[t] = a; ssq[t] = q;
  __syncthreads();
  for (int s = 128; s > 0; s >>= 1) {
    if (t < s) { ssum[t] += ssum[t + s]; ssq[t] += ssq[t + s]; }
    __syncthreads();
  }
  if (t == 0) {
    const float cnt = (float)B_ * (float)N_ * (float)N_;
    float mean = ssum[0] / cnt;
    float var  = ssq[0] / cnt - mean * mean;
    float sc   = gamma[g] * rsqrtf(var + EPS_);
    stats[g * 2 + 0] = sc;
    stats[g * 2 + 1] = beta[g] - mean * sc;
  }
}

// ---------------------------------------------------------------------------
// K5 (fused): BN-normalize attn in place AND compute attn@V with WMMA.
// One wave per (b,h,n16): the 32 lanes' A-fragment footprint covers each 16x32
// f32 chunk exactly once (row=lane&15; cols 8*hi+0..7 and 16+8*hi+0..7), so
// each lane normalizes its 16 values, stores them back to d_out, and converts
// them to the f16 A-fragment. Saves a full 402 MB read+write pass vs a
// standalone bnorm kernel.
// ---------------------------------------------------------------------------
__global__ void k_av_norm(float* __restrict__ attn,
                          const _Float16* __restrict__ Vt,
                          const float* __restrict__ stats,
                          _Float16* __restrict__ Y) {
  int blk = blockIdx.x;
  int n16 = blk & 63, bh = blk >> 6;
  int lane = threadIdx.x, hi = lane >> 4, col = lane & 15;
  int b = bh / NH_, h = bh % NH_;
  float sc = stats[h * 2 + 0], sh = stats[h * 2 + 1];

  float* ab = attn + ((size_t)bh * N_ + n16 * 16) * N_;
  const _Float16* vb = Vt + (size_t)bh * HD_ * N_;

  v8f acc[4];
#pragma unroll
  for (int dt = 0; dt < 4; ++dt) acc[dt] = (v8f){};

  for (int mc = 0; mc < 32; ++mc) {
    float* p = ab + (size_t)(lane & 15) * N_ + mc * 32 + 8 * hi;
    // load 2x8 contiguous f32, normalize, write back, pack to f16 A-frag
    float4 x0 = *(float4*)(p + 0);
    float4 x1 = *(float4*)(p + 4);
    float4 x2 = *(float4*)(p + 16);
    float4 x3 = *(float4*)(p + 20);
    x0.x = fmaf(x0.x, sc, sh); x0.y = fmaf(x0.y, sc, sh);
    x0.z = fmaf(x0.z, sc, sh); x0.w = fmaf(x0.w, sc, sh);
    x1.x = fmaf(x1.x, sc, sh); x1.y = fmaf(x1.y, sc, sh);
    x1.z = fmaf(x1.z, sc, sh); x1.w = fmaf(x1.w, sc, sh);
    x2.x = fmaf(x2.x, sc, sh); x2.y = fmaf(x2.y, sc, sh);
    x2.z = fmaf(x2.z, sc, sh); x2.w = fmaf(x2.w, sc, sh);
    x3.x = fmaf(x3.x, sc, sh); x3.y = fmaf(x3.y, sc, sh);
    x3.z = fmaf(x3.z, sc, sh); x3.w = fmaf(x3.w, sc, sh);
    *(float4*)(p + 0)  = x0;
    *(float4*)(p + 4)  = x1;
    *(float4*)(p + 16) = x2;
    *(float4*)(p + 20) = x3;

    H16 fr;
    fr.e[0]  = (_Float16)x0.x; fr.e[1]  = (_Float16)x0.y;
    fr.e[2]  = (_Float16)x0.z; fr.e[3]  = (_Float16)x0.w;
    fr.e[4]  = (_Float16)x1.x; fr.e[5]  = (_Float16)x1.y;
    fr.e[6]  = (_Float16)x1.z; fr.e[7]  = (_Float16)x1.w;
    fr.e[8]  = (_Float16)x2.x; fr.e[9]  = (_Float16)x2.y;
    fr.e[10] = (_Float16)x2.z; fr.e[11] = (_Float16)x2.w;
    fr.e[12] = (_Float16)x3.x; fr.e[13] = (_Float16)x3.y;
    fr.e[14] = (_Float16)x3.z; fr.e[15] = (_Float16)x3.w;

#pragma unroll
    for (int dt = 0; dt < 4; ++dt) {
      v16h bf = load_frag_bt16(vb + (size_t)(dt * 16) * N_ + mc * 32, N_, lane);
      acc[dt] = wmma_f16(fr.v, bf, acc[dt]);
    }
  }

#pragma unroll
  for (int dt = 0; dt < 4; ++dt) {
    F8 c; c.v = acc[dt];
#pragma unroll
    for (int r = 0; r < 8; ++r) {
      size_t t = (size_t)b * N_ + n16 * 16 + r + 8 * hi;
      Y[t * C_ + h * 64 + dt * 16 + col] = (_Float16)c.e[r];
    }
  }
}

// ---------------------------------------------------------------------------
// K6: projection GEMM: x[t,j] = sum_c Y[t,c]*W[j,c] + pb[j]
// one wave per 16x16 output tile: (4096/16)*(768/16) = 12288 blocks
// ---------------------------------------------------------------------------
__global__ void k_proj(const _Float16* __restrict__ Y,
                       const _Float16* __restrict__ W,
                       const float* __restrict__ pb,
                       float* __restrict__ outx) {
  int blk = blockIdx.x;
  int j16 = blk % 48, t16 = blk / 48;
  int lane = threadIdx.x, hi = lane >> 4, col = lane & 15;
  const _Float16* ab = Y + (size_t)t16 * 16 * C_;
  const _Float16* bb = W + (size_t)j16 * 16 * C_;
  v8f acc = (v8f){};
  for (int kc = 0; kc < 24; ++kc) {
    v16h a = load_frag_a16(ab + kc * 32, C_, lane);
    v16h bf = load_frag_bt16(bb + kc * 32, C_, lane);
    acc = wmma_f16(a, bf, acc);
  }
  float bias = pb[j16 * 16 + col];
  F8 c; c.v = acc;
#pragma unroll
  for (int r = 0; r < 8; ++r) {
    size_t t = (size_t)t16 * 16 + r + 8 * hi;
    outx[t * C_ + j16 * 16 + col] = c.e[r] + bias;
  }
}

// ---------------------------------------------------------------------------
extern "C" void kernel_launch(void* const* d_in, const int* in_sizes, int n_in,
                              void* d_out, int out_size, void* d_ws, size_t ws_size,
                              hipStream_t stream) {
  const float* X     = (const float*)d_in[0];
  const float* wq    = (const float*)d_in[1];
  const float* wk    = (const float*)d_in[2];
  const float* wv    = (const float*)d_in[3];
  const float* rw    = (const float*)d_in[4];
  const float* rb    = (const float*)d_in[5];
  const float* gamma = (const float*)d_in[6];
  const float* beta  = (const float*)d_in[7];
  const float* pw    = (const float*)d_in[8];
  const float* pb    = (const float*)d_in[9];

  float* out_x    = (float*)d_out;
  float* out_attn = out_x + (size_t)B_ * N_ * C_;

  // workspace carve-up (all 256B-aligned sizes; total ~28.3 MB)
  char* w = (char*)d_ws;
  _Float16* Q16  = (_Float16*)w; w += (size_t)B_ * NH_ * N_ * HD_ * 2;
  _Float16* K16  = (_Float16*)w; w += (size_t)B_ * NH_ * N_ * HD_ * 2;
  _Float16* Vt16 = (_Float16*)w; w += (size_t)B_ * NH_ * N_ * HD_ * 2;
  _Float16* Y16  = (_Float16*)w; w += (size_t)B_ * N_ * C_ * 2;
  _Float16* W16  = (_Float16*)w; w += (size_t)C_ * C_ * 2;
  float* rowmax  = (float*)w;    w += (size_t)B_ * NH_ * N_ * 4;
  float* rowinv  = (float*)w;    w += (size_t)B_ * NH_ * N_ * 4;
  float* partial = (float*)w;    w += (size_t)NH_ * 2 * 16384 * 4;
  float* stats   = (float*)w;    w += 256;

  k_w16cast <<<(C_ * C_ + 255) / 256, 256, 0, stream>>>(pw, W16);
  k_conv_qkv<<<B_ * N_,            256, 0, stream>>>(X, wq, wk, wv, Q16, K16, Vt16);
  k_rowstats<<<B_ * NH_ * (N_/16),  32, 0, stream>>>(Q16, K16, rowmax, rowinv);
  k_mix     <<<B_ * (N_/16) * (N_/16), 32, 0, stream>>>(Q16, K16, rowmax, rowinv,
                                                        rw, rb, out_attn, partial);
  k_redstats<<<NH_,                256, 0, stream>>>(partial, gamma, beta, stats);
  k_av_norm <<<B_ * NH_ * (N_/16),  32, 0, stream>>>(out_attn, Vt16, stats, Y16);
  k_proj    <<<(B_ * N_ / 16) * (C_ / 16), 32, 0, stream>>>(Y16, W16, pb, out_x);
}
